// Decoder_PointTrans_84086869721122
// MI455X (gfx1250) — matrix operand order repaired and checked by reading, not compiled, not run on hardware
//
#include <hip/hip_runtime.h>
#include <hip/hip_bf16.h>

// ---------------------------------------------------------------------------
// CDNA5 (gfx1250) Point-Transformer decoder.
// All linear layers run through one bf16 WMMA GEMM kernel
// (v_wmma_f32_16x16x32_bf16, f32 accumulate), register-blocked 2x4 tiles per
// wave (8 WMMAs per K-step), explicit global-address-space float4 loads
// batched ahead of the conversions so each K-step waits once. The [N,16,512]
// neighborhood pipeline is processed in 512-point chunks so its three 16MB
// scratch buffers stay resident in MI455X's 192MB L2.
// ---------------------------------------------------------------------------

typedef __attribute__((ext_vector_type(16))) __bf16 v16bf;
typedef __attribute__((ext_vector_type(8)))  float  v8f;
typedef __attribute__((ext_vector_type(4)))  float  v4f;

#define AS1 __attribute__((address_space(1)))

#define DM 512
#define KNN 16
#define CHUNK 512   // points per neighborhood-pipeline chunk

// ---------------------------------------------------------------------------
// GEMM: C[M,Nd] = A[M,K] * W[Nd,K]^T (+bias) (+residual) (opt ReLU)
// One wave computes an MT x NT grid of 16x16 tiles; K stepped by 32.
// A fragment layout (16-bit A 16x32): lane L (m=L&15, half=L>>4):
//   elems 0..7  -> K = kb + half*8 + t      (float4 pair at +0)
//   elems 8..15 -> K = kb + 16 + half*8 + t (float4 pair at +16 floats)
// B fragment layout (16-bit B 32x16): lane L (n=L&15, half=L>>4):
//   elems 0..7 -> K = kb + half*8 + t, 8..15 -> K = kb + 16 + half*8 + t (as A)
// C layout (v8f): row = m0 + half*8 + r, col = n0 + (L&15)
// ---------------------------------------------------------------------------
template <int MT, int NT>
__global__ void gemm_wmma_kernel(const float* __restrict__ A,
                                 const float* __restrict__ W,
                                 const float* __restrict__ bias,
                                 const float* __restrict__ resid,
                                 float* __restrict__ C,
                                 int M, int Nd, int K, int do_relu) {
  const int lane = threadIdx.x;           // 0..31
  const int m0 = (blockIdx.y * blockDim.y + threadIdx.y) * (16 * MT);
  const int n0 = blockIdx.x * (16 * NT);
  if (m0 >= M) return;                     // whole-wave uniform: EXEC stays all-1
  const int half = lane >> 4;
  const int l15  = lane & 15;

  // explicit global address space: avoid FLAT path (flat ticks DScnt too)
  const AS1 float* Ag = (const AS1 float*)A;
  const AS1 float* Wg = (const AS1 float*)W;

  const AS1 v4f* ap[MT];   // points at K = kb + half*8 within row
  const AS1 v4f* wp[NT];   // points at K = kb + half*8 within row
#pragma unroll
  for (int mt = 0; mt < MT; ++mt)
    ap[mt] = (const AS1 v4f*)(Ag + (size_t)(m0 + mt * 16 + l15) * K + half * 8);
#pragma unroll
  for (int nt = 0; nt < NT; ++nt)
    wp[nt] = (const AS1 v4f*)(Wg + (size_t)(n0 + nt * 16 + l15) * K + half * 8);

  v8f acc[MT][NT];
#pragma unroll
  for (int mt = 0; mt < MT; ++mt)
#pragma unroll
    for (int nt = 0; nt < NT; ++nt) acc[mt][nt] = {};

#pragma unroll 1
  for (int kb = 0; kb < K; kb += 32) {
    // ---- phase 1: issue ALL loads of this K-step into distinct temps ----
    v4f at[MT][4];   // [0],[1] = run at +0 ; [2],[3] = run at +16 floats
    v4f bt[NT][4];
#pragma unroll
    for (int mt = 0; mt < MT; ++mt) {
      at[mt][0] = ap[mt][0];
      at[mt][1] = ap[mt][1];
      at[mt][2] = ap[mt][4];
      at[mt][3] = ap[mt][5];
      ap[mt] += 8;           // +32 floats
    }
#pragma unroll
    for (int nt = 0; nt < NT; ++nt) {
      bt[nt][0] = wp[nt][0];
      bt[nt][1] = wp[nt][1];
      bt[nt][2] = wp[nt][4];
      bt[nt][3] = wp[nt][5];
      wp[nt] += 8;           // +32 floats
    }
    // ---- phase 2: convert to bf16 fragments ----
    v16bf a[MT], b[NT];
#pragma unroll
    for (int mt = 0; mt < MT; ++mt)
#pragma unroll
      for (int t = 0; t < 16; ++t) a[mt][t] = (__bf16)at[mt][t >> 2][t & 3];
#pragma unroll
    for (int nt = 0; nt < NT; ++nt)
#pragma unroll
      for (int t = 0; t < 16; ++t) b[nt][t] = (__bf16)bt[nt][t >> 2][t & 3];
    // ---- phase 3: 8 WMMAs ----
#pragma unroll
    for (int nt = 0; nt < NT; ++nt)
#pragma unroll
      for (int mt = 0; mt < MT; ++mt)
        acc[mt][nt] = __builtin_amdgcn_wmma_f32_16x16x32_bf16(
            false, a[mt], false, b[nt], (short)0, acc[mt][nt], false, false);
  }

  AS1 float* Cg = (AS1 float*)C;
  const AS1 float* Rg = (const AS1 float*)resid;
  const AS1 float* Bg = (const AS1 float*)bias;
#pragma unroll
  for (int nt = 0; nt < NT; ++nt) {
    const int col = n0 + nt * 16 + l15;
    const float bv = bias ? Bg[col] : 0.0f;
#pragma unroll
    for (int mt = 0; mt < MT; ++mt) {
      const int r0 = m0 + mt * 16 + half * 8;
#pragma unroll
      for (int r = 0; r < 8; ++r) {
        size_t off = (size_t)(r0 + r) * Nd + col;
        float v = acc[mt][nt][r] + bv;
        if (resid) v += Rg[off];
        if (do_relu) v = fmaxf(v, 0.0f);
        Cg[off] = v;
      }
    }
  }
}

// ---------------------------------------------------------------------------
// BatchNorm1d (training-mode biased stats over rows) + ReLU, one block/channel.
// ---------------------------------------------------------------------------
__global__ void bn_relu_kernel(const float* __restrict__ x, float* __restrict__ y,
                               const float* __restrict__ g, const float* __restrict__ b,
                               int Nrows, int C) {
  const int c = blockIdx.x;
  __shared__ float ssum[256], ssq[256];
  float s = 0.0f, q = 0.0f;
  for (int r = threadIdx.x; r < Nrows; r += blockDim.x) {
    float v = x[(size_t)r * C + c];
    s += v; q += v * v;
  }
  ssum[threadIdx.x] = s; ssq[threadIdx.x] = q;
  __syncthreads();
  for (int st = 128; st > 0; st >>= 1) {
    if (threadIdx.x < st) {
      ssum[threadIdx.x] += ssum[threadIdx.x + st];
      ssq[threadIdx.x]  += ssq[threadIdx.x + st];
    }
    __syncthreads();
  }
  const float mean = ssum[0] / (float)Nrows;
  const float var  = ssq[0] / (float)Nrows - mean * mean;
  const float scale = g[c] * rsqrtf(var + 1e-5f);
  const float shift = b[c] - mean * scale;
  for (int r = threadIdx.x; r < Nrows; r += blockDim.x) {
    float v = x[(size_t)r * C + c] * scale + shift;
    y[(size_t)r * C + c] = fmaxf(v, 0.0f);
  }
}

// ---------------------------------------------------------------------------
// KNN-16 (includes self), one thread per query point, insertion into top-16.
// ---------------------------------------------------------------------------
__global__ void knn16_kernel(const float* __restrict__ xyz, int N, int* __restrict__ idx) {
  const int i = blockIdx.x * blockDim.x + threadIdx.x;
  if (i >= N) return;
  const float xi = xyz[3 * i], yi = xyz[3 * i + 1], zi = xyz[3 * i + 2];
  float bd[KNN]; int bi[KNN];
#pragma unroll
  for (int t = 0; t < KNN; ++t) { bd[t] = 3.0e38f; bi[t] = 0; }
  for (int j = 0; j < N; ++j) {
    float dx = xi - xyz[3 * j], dy = yi - xyz[3 * j + 1], dz = zi - xyz[3 * j + 2];
    float d = dx * dx + dy * dy + dz * dz;
    if (d < bd[KNN - 1]) {
      int p = KNN - 1;
      while (p > 0 && bd[p - 1] > d) { bd[p] = bd[p - 1]; bi[p] = bi[p - 1]; --p; }
      bd[p] = d; bi[p] = j;
    }
  }
#pragma unroll
  for (int t = 0; t < KNN; ++t) idx[(size_t)i * KNN + t] = bi[t];
}

// ---------------------------------------------------------------------------
// 3-NN inverse-distance feature propagation: out = interp(feat_c) + addv.
// ---------------------------------------------------------------------------
__global__ void interp3_kernel(const float* __restrict__ xf, int Nf,
                               const float* __restrict__ xc, int Nc,
                               const float* __restrict__ featc, int C,
                               const float* __restrict__ addv,
                               float* __restrict__ out) {
  const int i = blockIdx.x * blockDim.x + threadIdx.x;
  if (i >= Nf) return;
  const float px = xf[3 * i], py = xf[3 * i + 1], pz = xf[3 * i + 2];
  float d0 = 3.0e38f, d1 = 3.0e38f, d2 = 3.0e38f;
  int i0 = 0, i1 = 0, i2 = 0;
  for (int j = 0; j < Nc; ++j) {
    float dx = px - xc[3 * j], dy = py - xc[3 * j + 1], dz = pz - xc[3 * j + 2];
    float d = dx * dx + dy * dy + dz * dz;
    if (d < d0)      { d2 = d1; i2 = i1; d1 = d0; i1 = i0; d0 = d; i0 = j; }
    else if (d < d1) { d2 = d1; i2 = i1; d1 = d; i1 = j; }
    else if (d < d2) { d2 = d; i2 = j; }
  }
  float w0 = 1.0f / (d0 + 1e-8f), w1 = 1.0f / (d1 + 1e-8f), w2 = 1.0f / (d2 + 1e-8f);
  const float inv = 1.0f / (w0 + w1 + w2);
  w0 *= inv; w1 *= inv; w2 *= inv;
  for (int c = 0; c < C; ++c) {
    out[(size_t)i * C + c] = w0 * featc[(size_t)i0 * C + c]
                           + w1 * featc[(size_t)i1 * C + c]
                           + w2 * featc[(size_t)i2 * C + c]
                           + addv[(size_t)i * C + c];
  }
}

// ---------------------------------------------------------------------------
// pos-enc first layer: h = relu(W1 * (xyz[n]-xyz[knn[n,j]]) + b1)  (K=3)
// ---------------------------------------------------------------------------
__global__ void posenc_h_kernel(const float* __restrict__ xyz, const int* __restrict__ knn,
                                const float* __restrict__ w1, const float* __restrict__ b1,
                                float* __restrict__ h, int nBase, int total) {
  const int lin = blockIdx.x * blockDim.x + threadIdx.x;
  if (lin >= total) return;
  const int c = lin & (DM - 1);
  const int row = lin >> 9;                // nl*16 + j
  const int nl = row >> 4, j = row & 15;
  const int n = nBase + nl;
  const int nb = knn[(size_t)n * KNN + j];
  const float dx = xyz[3 * n]     - xyz[3 * nb];
  const float dy = xyz[3 * n + 1] - xyz[3 * nb + 1];
  const float dz = xyz[3 * n + 2] - xyz[3 * nb + 2];
  const float v = w1[c * 3] * dx + w1[c * 3 + 1] * dy + w1[c * 3 + 2] * dz + b1[c];
  h[lin] = fmaxf(v, 0.0f);
}

// attn_in = q[n] - k[knn[n,j]] + pos[n,j]
__global__ void attn_in_kernel(const float* __restrict__ q, const float* __restrict__ kx,
                               const int* __restrict__ knn, const float* __restrict__ pos,
                               float* __restrict__ out, int nBase, int total) {
  const int lin = blockIdx.x * blockDim.x + threadIdx.x;
  if (lin >= total) return;
  const int c = lin & (DM - 1);
  const int row = lin >> 9;
  const int nl = row >> 4, j = row & 15;
  const int n = nBase + nl;
  const int nb = knn[(size_t)n * KNN + j];
  out[lin] = q[(size_t)n * DM + c] - kx[(size_t)nb * DM + c] + pos[lin];
}

// softmax over the 16 neighbors (axis=-2), scaled by 1/sqrt(512); in-place.
__global__ void softmax_kernel(float* __restrict__ a, int total /* = chunkPts*512 */) {
  const int lin = blockIdx.x * blockDim.x + threadIdx.x;
  if (lin >= total) return;
  const int c = lin & (DM - 1);
  const int nl = lin >> 9;
  const float scale = 0.04419417382415922f;  // 1/sqrt(512)
  const size_t base = ((size_t)nl * KNN) * DM + c;
  float vals[KNN];
  float m = -3.0e38f;
#pragma unroll
  for (int j = 0; j < KNN; ++j) {
    float v = a[base + (size_t)j * DM] * scale;
    vals[j] = v;
    m = fmaxf(m, v);
  }
  float s = 0.0f;
#pragma unroll
  for (int j = 0; j < KNN; ++j) { vals[j] = __expf(vals[j] - m); s += vals[j]; }
  const float inv = 1.0f / s;
#pragma unroll
  for (int j = 0; j < KNN; ++j) a[base + (size_t)j * DM] = vals[j] * inv;
}

// res[n,c] = sum_j attn[n,j,c] * (v[knn[n,j],c] + pos[n,j,c])
__global__ void attn_reduce_kernel(const float* __restrict__ a, const float* __restrict__ vx,
                                   const int* __restrict__ knn, const float* __restrict__ pos,
                                   float* __restrict__ res, int nBase, int total) {
  const int lin = blockIdx.x * blockDim.x + threadIdx.x;
  if (lin >= total) return;
  const int c = lin & (DM - 1);
  const int nl = lin >> 9;
  const int n = nBase + nl;
  const size_t base = ((size_t)nl * KNN) * DM + c;
  float s = 0.0f;
#pragma unroll
  for (int j = 0; j < KNN; ++j) {
    const int nb = knn[(size_t)n * KNN + j];
    s += a[base + (size_t)j * DM] * (vx[(size_t)nb * DM + c] + pos[base + (size_t)j * DM]);
  }
  res[(size_t)n * DM + c] = s;
}

// ---------------------------------------------------------------------------
// Host side
// ---------------------------------------------------------------------------
struct TFp {
  const float *d1b, *d1w, *d2b, *d2w, *fc1b, *fc1w, *fc2b, *fc2w;
  const float *g1b, *g1w, *g2b, *g2w, *wk, *wq, *wv;
};
struct TUp {
  const float *bn1b, *bn1g, *bn2b, *bn2g, *fc1b, *fc1w, *fc2b, *fc2w;
};

static inline void gemm(hipStream_t s, const float* A, const float* W, const float* bias,
                        const float* resid, float* C, int M, int Nd, int K, bool relu) {
  const dim3 block(32, 4);
  // MT=2: each wave covers 32 rows; 4 waves/block -> 128 rows/block.
  const int gy = (M + 127) / 128;
  if (Nd % 64 == 0) {
    dim3 grid(Nd / 64, gy);
    gemm_wmma_kernel<2, 4><<<grid, block, 0, s>>>(A, W, bias, resid, C, M, Nd, K, relu ? 1 : 0);
  } else {
    dim3 grid(Nd / 32, gy);
    gemm_wmma_kernel<2, 2><<<grid, block, 0, s>>>(A, W, bias, resid, C, M, Nd, K, relu ? 1 : 0);
  }
}

extern "C" void kernel_launch(void* const* d_in, const int* in_sizes, int n_in,
                              void* d_out, int out_size, void* d_ws, size_t ws_size,
                              hipStream_t stream) {
  (void)in_sizes; (void)n_in; (void)out_size; (void)ws_size;

  const float* xyz    = (const float*)d_in[0];   // [256,3]
  const float* points = (const float*)d_in[1];   // [256,128]
  const float* xyz0   = (const float*)d_in[2];   // [4096,3]
  const float* f0     = (const float*)d_in[3];   // [4096,32]
  const float* xyz1   = (const float*)d_in[4];   // [1024,3]
  const float* f1     = (const float*)d_in[5];   // [1024,64]

  // params flattened in JAX pytree order (dict keys sorted at every level):
  // tf0, tf1, tu0, tu1; within tf: d1(b,w) d2(b,w) fc1(b,w) fc2(b,w) g1(b,w)
  // g2(b,w) wk.w wq.w wv.w ; within tu: bn1(b,g) bn2(b,g) fc1(b,w) fc2(b,w)
  auto P = [&](int i) { return (const float*)d_in[i]; };
  TFp tf0{P(6),  P(7),  P(8),  P(9),  P(10), P(11), P(12), P(13),
          P(14), P(15), P(16), P(17), P(18), P(19), P(20)};
  TFp tf1{P(21), P(22), P(23), P(24), P(25), P(26), P(27), P(28),
          P(29), P(30), P(31), P(32), P(33), P(34), P(35)};
  TUp tu0{P(36), P(37), P(38), P(39), P(40), P(41), P(42), P(43)};
  TUp tu1{P(44), P(45), P(46), P(47), P(48), P(49), P(50), P(51)};

  // Workspace layout (bytes), max sizes for N=4096 / CHUNK=512.
  char* ws = (char*)d_ws;
  int*   IDX  = (int*)  (ws + 0);          // 4096*16*4           = 256 KB
  float* TMPC = (float*)(ws + 262144);     // coarse linear out   = 128 KB
  float* FINT = (float*)(ws + 393216);     // fine linear out     = 512 KB
  float* PTS0 = (float*)(ws + 917504);     // block-0 points      = 256 KB
  float* PTS1 = (float*)(ws + 1179648);    // block-1 points_new  = 512 KB
  float* Xb   = (float*)(ws + 1703936);    // [4096,512] 8 MB
  float* Qb   = (float*)(ws + 10092544);   // 8 MB
  float* KXb  = (float*)(ws + 18481152);   // 8 MB
  float* VXb  = (float*)(ws + 26869760);   // 8 MB
  float* RESb = (float*)(ws + 35258368);   // 8 MB
  float* POSb = (float*)(ws + 43646976);   // [8192,512] 16 MB (L2-resident)
  float* BUFA = (float*)(ws + 60424192);   // 16 MB
  float* BUFB = (float*)(ws + 77201408);   // 16 MB

  auto runBlock = [&](const float* cxyz, int Nc, const float* cpts, int ch2,
                      const float* fxyz, int Nf, const float* ff, int ch,
                      const TUp& tu, const TFp& tf, float* ptsNew, float* outPts) {
    // --- transition_up ---
    gemm(stream, cpts, tu.fc1w, tu.fc1b, nullptr, TMPC, Nc, ch, ch2, false);
    bn_relu_kernel<<<ch, 256, 0, stream>>>(TMPC, TMPC, tu.bn1g, tu.bn1b, Nc, ch);
    gemm(stream, ff, tu.fc2w, tu.fc2b, nullptr, FINT, Nf, ch, ch, false);
    bn_relu_kernel<<<ch, 256, 0, stream>>>(FINT, FINT, tu.bn2g, tu.bn2b, Nf, ch);
    interp3_kernel<<<(Nf + 255) / 256, 256, 0, stream>>>(fxyz, Nf, cxyz, Nc, TMPC, ch, FINT, ptsNew);

    // --- transformer block ---
    knn16_kernel<<<(Nf + 255) / 256, 256, 0, stream>>>(fxyz, Nf, IDX);
    gemm(stream, ptsNew, tf.fc1w, tf.fc1b, nullptr, Xb, Nf, DM, ch, false);
    gemm(stream, Xb, tf.wq, nullptr, nullptr, Qb,  Nf, DM, DM, false);
    gemm(stream, Xb, tf.wk, nullptr, nullptr, KXb, Nf, DM, DM, false);
    gemm(stream, Xb, tf.wv, nullptr, nullptr, VXb, Nf, DM, DM, false);

    for (int nb = 0; nb < Nf; nb += CHUNK) {
      const int rows = CHUNK * KNN;          // 8192
      const int tot  = rows * DM;            // 4.19M elements
      posenc_h_kernel<<<tot / 256, 256, 0, stream>>>(fxyz, IDX, tf.d1w, tf.d1b, BUFA, nb, tot);
      gemm(stream, BUFA, tf.d2w, tf.d2b, nullptr, POSb, rows, DM, DM, false);
      attn_in_kernel<<<tot / 256, 256, 0, stream>>>(Qb, KXb, IDX, POSb, BUFA, nb, tot);
      gemm(stream, BUFA, tf.g1w, tf.g1b, nullptr, BUFB, rows, DM, DM, true);
      gemm(stream, BUFB, tf.g2w, tf.g2b, nullptr, BUFA, rows, DM, DM, false);
      const int tot2 = CHUNK * DM;
      softmax_kernel<<<tot2 / 256, 256, 0, stream>>>(BUFA, tot2);
      attn_reduce_kernel<<<tot2 / 256, 256, 0, stream>>>(BUFA, VXb, IDX, POSb, RESb, nb, tot2);
    }
    // fc2 + residual (residual may alias output: each element read-then-written once)
    gemm(stream, RESb, tf.fc2w, tf.fc2b, ptsNew, outPts, Nf, ch, DM, false);
  };

  float* outF = (float*)d_out;
  // block i=0: coarse(xyz,points ch2=128) -> fine(xyz1,f1 ch=64)
  runBlock(xyz, 256, points, 128, xyz1, 1024, f1, 64, tu0, tf0, PTS0, PTS0);
  // block i=1: coarse(xyz1, PTS0 ch2=64) -> fine(xyz0,f0 ch=32), writes d_out points
  runBlock(xyz1, 1024, PTS0, 64, xyz0, 4096, f0, 32, tu1, tf1, PTS1, outF + 12288);
  // output tuple: (xyz0, points_final)
  hipMemcpyAsync(outF, xyz0, 12288 * sizeof(float), hipMemcpyDeviceToDevice, stream);
}
